// SEAL_20203526160739
// MI455X (gfx1250) — matrix operand order repaired and checked by reading, not compiled, not run on hardware
//
#include <hip/hip_runtime.h>

typedef __attribute__((ext_vector_type(16))) _Float16 v16h;
typedef __attribute__((ext_vector_type(8)))  float    v8f;

// ---------------------------------------------------------------------------
// WMMA GEMM:  Y[M x NC] = X[M x K] @ W[K x NC]  (+bias) using f16 WMMA, f32 acc.
// NCP = NC padded to multiple of 16 (padding columns zeroed in LDS).
// SEG: atomicAdd row into Y[seg[row]*NC + n] instead of storing (fused pool).
// Block = 256 threads = 8 waves; each wave owns 16 rows; 128 rows/block.
//
// A fragment (ISA 7.12.2, 16-bit A 16x32): lane (m16,grp) holds
//   a[0..7]  = X[row][kb + grp*8 + 0..7]       -> one float4 pair
//   a[8..15] = X[row][kb + 16 + grp*8 + 0..7]  -> one float4 pair
// Rows are clamped (not predicated): invalid rows produce garbage accumulator
// rows that are discarded by the store guard, so loads stay branch-free.
//
// B fragment (32x16): lane holds halves h=0..15 -> W[kb + grp*8 + (h&7) + 16*(h>>3)][n].
// W is staged in LDS pre-swizzled so each lane's fragment is 16 contiguous
// halves (32B aligned) => 2 x ds_load_b128 per fragment.
// ---------------------------------------------------------------------------
template<int K, int NCP, int NC, bool HAS_BIAS, bool SEG>
__global__ __launch_bounds__(256)
void gemm_wmma_k(const float* __restrict__ X, const float* __restrict__ W,
                 const float* __restrict__ bias, const int* __restrict__ seg,
                 float* __restrict__ Y, int M)
{
    constexpr int NT = NCP / 16;   // N tiles per wave
    constexpr int KB = K / 32;     // K blocks
    __shared__ _Float16 Ws[K * NCP];   // fragment-order layout

    const int tid = threadIdx.x;
    // Stage W swizzled: linear dest i -> (kbi, t, grp, nn, h)
    for (int i = tid; i < K * NCP; i += 256) {
        const int h   = i & 15;
        const int r1  = i >> 4;
        const int nn  = r1 & 15;
        const int r2  = r1 >> 4;
        const int grp = r2 & 1;
        const int r3  = r2 >> 1;
        const int t   = r3 % NT;
        const int kbi = r3 / NT;
        const int k   = kbi * 32 + grp * 8 + (h & 7) + ((h >> 3) << 4);
        const int n   = t * 16 + nn;
        Ws[i] = (n < NC) ? (_Float16)W[k * NC + n] : (_Float16)0.0f;
    }
    __syncthreads();

    const int wave = tid >> 5;
    const int lane = tid & 31;
    const int m16  = lane & 15;
    const int grp  = lane >> 4;
    const int r0   = blockIdx.x * 128 + wave * 16;

    v8f acc[NT] = {};

    const int rowA   = min(r0 + m16, M - 1);       // clamp: branch-free loads
    const float* xr  = X + (size_t)rowA * K;

#pragma unroll
    for (int kbi = 0; kbi < KB; ++kbi) {
        const float* xp = xr + kbi * 32 + grp * 8;
        const float4 q0 = *(const float4*)(xp + 0);
        const float4 q1 = *(const float4*)(xp + 4);
        const float4 q2 = *(const float4*)(xp + 16);
        const float4 q3 = *(const float4*)(xp + 20);
        v16h a;
        a[0]  = (_Float16)q0.x; a[1]  = (_Float16)q0.y;
        a[2]  = (_Float16)q0.z; a[3]  = (_Float16)q0.w;
        a[4]  = (_Float16)q1.x; a[5]  = (_Float16)q1.y;
        a[6]  = (_Float16)q1.z; a[7]  = (_Float16)q1.w;
        a[8]  = (_Float16)q2.x; a[9]  = (_Float16)q2.y;
        a[10] = (_Float16)q2.z; a[11] = (_Float16)q2.w;
        a[12] = (_Float16)q3.x; a[13] = (_Float16)q3.y;
        a[14] = (_Float16)q3.z; a[15] = (_Float16)q3.w;
#pragma unroll
        for (int t = 0; t < NT; ++t) {
            const v16h b = *(const v16h*)(Ws + ((size_t)((kbi * NT + t) * 32 + lane) * 16));
            acc[t] = __builtin_amdgcn_wmma_f32_16x16x32_f16(
                false, a, false, b, (short)0, acc[t], false, false);
        }
    }

    // Hoist segment ids (SEG path) once for the wave's 8 rows per lane group.
    int segr[8];
    if (SEG) {
#pragma unroll
        for (int v = 0; v < 8; ++v) {
            const int row = r0 + v + 8 * grp;
            segr[v] = (row < M) ? seg[row] : 0;
        }
    }

    // C/D layout: VGPR v, lane -> (M = r0 + v + 8*grp, N = t*16 + m16)
#pragma unroll
    for (int t = 0; t < NT; ++t) {
        const int n = t * 16 + m16;
        if (NC < NCP && n >= NC) continue;   // padded columns
#pragma unroll
        for (int v = 0; v < 8; ++v) {
            const int row = r0 + v + 8 * grp;
            if (row >= M) continue;
            float val = acc[t][v];
            if (HAS_BIAS) val += bias[n];
            if (SEG) {
                atomicAdd(&Y[(size_t)segr[v] * NC + n], val);
            } else {
                Y[(size_t)row * NC + n] = val;
            }
        }
    }
}

// ---------------------------------------------------------------------------
// Elementwise / scatter helpers (32-bit index math, compile-time feature dims)
// ---------------------------------------------------------------------------
__global__ void fill_k(float* __restrict__ p, float v, unsigned n)
{
    const unsigned t = blockIdx.x * blockDim.x + threadIdx.x;
    if (t < n) p[t] = v;
}

__global__ void scatter_count_k(const int* __restrict__ idx, float* __restrict__ c,
                                unsigned n)
{
    const unsigned t = blockIdx.x * blockDim.x + threadIdx.x;
    if (t < n) atomicAdd(&c[idx[t]], 1.0f);
}

// agg[i,f] = xw[i,f] / deg[i] + bias[f]   (self-loop term + layer bias)
template<int F>
__global__ void init_agg_k(const float* __restrict__ xw, const float* __restrict__ deg,
                           const float* __restrict__ bias, float* __restrict__ agg,
                           unsigned total)
{
    const unsigned t = blockIdx.x * blockDim.x + threadIdx.x;
    if (t >= total) return;
    const unsigned f = t % F;
    const unsigned i = t / F;
    agg[t] = xw[t] / deg[i] + bias[f];
}

// agg[dst] += xw[src] * rsqrt(deg[src]) * rsqrt(deg[dst]), float4 per thread
template<int F>
__global__ void scatter_edges4_k(const int* __restrict__ src, const int* __restrict__ dst,
                                 const float* __restrict__ deg, const float* __restrict__ xw,
                                 float* __restrict__ agg, unsigned total)
{
    constexpr unsigned CH = F >> 2;
    const unsigned t = blockIdx.x * blockDim.x + threadIdx.x;
    if (t >= total) return;
    const unsigned e = t / CH;
    const unsigned c = t % CH;
    const int s = src[e], d = dst[e];
    const float nrm = rsqrtf(deg[s]) * rsqrtf(deg[d]);
    const float4 xv = *(const float4*)(xw + (size_t)s * F + c * 4u);
    float* o = agg + (size_t)d * F + c * 4u;
    atomicAdd(o + 0, xv.x * nrm);
    atomicAdd(o + 1, xv.y * nrm);
    atomicAdd(o + 2, xv.z * nrm);
    atomicAdd(o + 3, xv.w * nrm);
}

// scalar variant (for F not divisible by 4, e.g. F = 10)
template<int F>
__global__ void scatter_edges1_k(const int* __restrict__ src, const int* __restrict__ dst,
                                 const float* __restrict__ deg, const float* __restrict__ xw,
                                 float* __restrict__ agg, unsigned total)
{
    const unsigned t = blockIdx.x * blockDim.x + threadIdx.x;
    if (t >= total) return;
    const unsigned e = t / F;
    const unsigned f = t % F;
    const int s = src[e], d = dst[e];
    const float nrm = rsqrtf(deg[s]) * rsqrtf(deg[d]);
    atomicAdd(&agg[(size_t)d * F + f], xw[(size_t)s * F + f] * nrm);
}

__global__ void relu4_k(float* __restrict__ p, unsigned n4)
{
    const unsigned t = blockIdx.x * blockDim.x + threadIdx.x;
    if (t >= n4) return;
    float4 v = ((float4*)p)[t];
    v.x = fmaxf(v.x, 0.0f); v.y = fmaxf(v.y, 0.0f);
    v.z = fmaxf(v.z, 0.0f); v.w = fmaxf(v.w, 0.0f);
    ((float4*)p)[t] = v;
}

// out = relu( (y - m) * rsqrt(v + eps) * g + be )
template<int D>
__global__ void bn_relu_k(const float* __restrict__ y, const float* __restrict__ g,
                          const float* __restrict__ be, const float* __restrict__ m,
                          const float* __restrict__ v, float* __restrict__ out,
                          unsigned total)
{
    const unsigned t = blockIdx.x * blockDim.x + threadIdx.x;
    if (t >= total) return;
    const unsigned f = t % D;
    const float z = (y[t] - m[f]) * rsqrtf(v[f] + 1e-5f) * g[f] + be[f];
    out[t] = fmaxf(z, 0.0f);
}

// loc[g,f] = sums[g,f] / max(cnt[g], 1)
template<int D>
__global__ void mean_div_k(const float* __restrict__ sums, const float* __restrict__ cnt,
                           float* __restrict__ loc, unsigned total)
{
    const unsigned t = blockIdx.x * blockDim.x + threadIdx.x;
    if (t >= total) return;
    const unsigned g = t / D;
    loc[t] = sums[t] / fmaxf(cnt[g], 1.0f);
}

// ---------------------------------------------------------------------------
// Host-side orchestration
// ---------------------------------------------------------------------------
static inline unsigned cdivu(unsigned long long a, unsigned long long b)
{
    return (unsigned)((a + b - 1) / b);
}

extern "C" void kernel_launch(void* const* d_in, const int* in_sizes, int n_in,
                              void* d_out, int out_size, void* d_ws, size_t ws_size,
                              hipStream_t stream)
{
    (void)n_in; (void)ws_size;
    // Fixed model dims per the reference; N/E/EM/G derived from sizes.
    const int      N  = in_sizes[0] / 64;
    const unsigned E  = (unsigned)(in_sizes[1] / 2);
    const unsigned EM = (unsigned)(in_sizes[3] / 2);
    const int      G  = (out_size - 1) / 276;   // out = G*(10+10+128+128)+1

    const float* x     = (const float*)d_in[0];
    const int*   ei    = (const int*)d_in[1];
    const int*   batch = (const int*)d_in[2];
    const int*   me    = (const int*)d_in[3];
    const float* W1  = (const float*)d_in[5],  *b1  = (const float*)d_in[6];
    const float* W2  = (const float*)d_in[7],  *b2  = (const float*)d_in[8];
    const float* FW1 = (const float*)d_in[9],  *Fb1 = (const float*)d_in[10];
    const float* FW2 = (const float*)d_in[11], *Fb2 = (const float*)d_in[12];
    const float* MW1 = (const float*)d_in[13], *mb1 = (const float*)d_in[14];
    const float* g1  = (const float*)d_in[15], *be1 = (const float*)d_in[16];
    const float* m1  = (const float*)d_in[17], *v1  = (const float*)d_in[18];
    const float* MW2 = (const float*)d_in[19], *mb2 = (const float*)d_in[20];
    const float* g2  = (const float*)d_in[21], *be2 = (const float*)d_in[22];
    const float* m2  = (const float*)d_in[23], *v2  = (const float*)d_in[24];
    const float* MW3 = (const float*)d_in[25], *mb3 = (const float*)d_in[26];

    const int* src = ei;            const int* dst = ei + E;
    const int* ms  = me;            const int* md  = me + EM;

    float* out  = (float*)d_out;
    float* hc   = out;                             // [G,10]
    float* ic   = out + (size_t)G * 10;            // [G,10]
    float* loc  = out + (size_t)G * 20;            // [G,128]
    float* glob = loc + (size_t)G * 128;           // [G,128]

    // Workspace carve (~280 MB)
    float* p     = (float*)d_ws;
    float* deg_n = p; p += N;
    float* degG  = p; p += G;
    float* cnt   = p; p += G;
    float* bufA  = p; p += (size_t)N * 64;         // xw
    float* bufB  = p; p += (size_t)N * 64;         // agg / h
    float* sums  = p; p += (size_t)G * 128;
    float* mxw   = p; p += (size_t)G * 128;
    float* magg  = p; p += (size_t)G * 128;
    float* mz    = p; p += (size_t)G * 128;
    float* hxw   = p; p += (size_t)G * 10;

    const int T = 256;
    const unsigned N64  = (unsigned)N * 64u;
    const unsigned G128 = (unsigned)G * 128u;
    const unsigned G10  = (unsigned)G * 10u;

    // --- degrees / counts --------------------------------------------------
    fill_k<<<cdivu(N, T), T, 0, stream>>>(deg_n, 1.0f, N);
    fill_k<<<cdivu(G, T), T, 0, stream>>>(degG, 1.0f, G);
    fill_k<<<cdivu(G, T), T, 0, stream>>>(cnt, 0.0f, G);
    fill_k<<<cdivu(G128, T), T, 0, stream>>>(sums, 0.0f, G128);
    scatter_count_k<<<cdivu(E, T), T, 0, stream>>>(dst, deg_n, E);
    scatter_count_k<<<cdivu(EM, T), T, 0, stream>>>(md, degG, EM);
    scatter_count_k<<<cdivu(N, T), T, 0, stream>>>(batch, cnt, N);

    // --- node GCN layer 1: h1 = relu(gcn(x, W1, b1)) -----------------------
    gemm_wmma_k<64, 64, 64, false, false><<<cdivu(N, 128), T, 0, stream>>>(
        x, W1, nullptr, nullptr, bufA, N);
    init_agg_k<64><<<cdivu(N64, T), T, 0, stream>>>(bufA, deg_n, b1, bufB, N64);
    scatter_edges4_k<64><<<cdivu(E * 16u, T), T, 0, stream>>>(
        src, dst, deg_n, bufA, bufB, E * 16u);
    relu4_k<<<cdivu(N64 / 4u, T), T, 0, stream>>>(bufB, N64 / 4u);

    // --- node GCN layer 2: h2 = relu(gcn(h1, W2, b2)) ----------------------
    gemm_wmma_k<64, 64, 64, false, false><<<cdivu(N, 128), T, 0, stream>>>(
        bufB, W2, nullptr, nullptr, bufA, N);
    init_agg_k<64><<<cdivu(N64, T), T, 0, stream>>>(bufA, deg_n, b2, bufB, N64);
    scatter_edges4_k<64><<<cdivu(E * 16u, T), T, 0, stream>>>(
        src, dst, deg_n, bufA, bufB, E * 16u);
    relu4_k<<<cdivu(N64 / 4u, T), T, 0, stream>>>(bufB, N64 / 4u);

    // --- a1 = h2 @ FW1 + Fb1, fused segment-sum pool into sums -------------
    gemm_wmma_k<64, 128, 128, true, true><<<cdivu(N, 128), T, 0, stream>>>(
        bufB, FW1, Fb1, batch, sums, N);
    mean_div_k<128><<<cdivu(G128, T), T, 0, stream>>>(sums, cnt, loc, G128);

    // --- ic = loc @ FW2 + Fb2 ---------------------------------------------
    gemm_wmma_k<128, 16, 10, true, false><<<cdivu(G, 128), T, 0, stream>>>(
        loc, FW2, Fb2, nullptr, ic, G);

    // --- macro GCN layer 1 -------------------------------------------------
    gemm_wmma_k<128, 128, 128, false, false><<<cdivu(G, 128), T, 0, stream>>>(
        loc, MW1, nullptr, nullptr, mxw, G);
    init_agg_k<128><<<cdivu(G128, T), T, 0, stream>>>(mxw, degG, mb1, magg, G128);
    scatter_edges4_k<128><<<cdivu(EM * 32u, T), T, 0, stream>>>(
        ms, md, degG, mxw, magg, EM * 32u);
    bn_relu_k<128><<<cdivu(G128, T), T, 0, stream>>>(magg, g1, be1, m1, v1, mz, G128);

    // --- macro GCN layer 2 -> glob ----------------------------------------
    gemm_wmma_k<128, 128, 128, false, false><<<cdivu(G, 128), T, 0, stream>>>(
        mz, MW2, nullptr, nullptr, mxw, G);
    init_agg_k<128><<<cdivu(G128, T), T, 0, stream>>>(mxw, degG, mb2, magg, G128);
    scatter_edges4_k<128><<<cdivu(EM * 32u, T), T, 0, stream>>>(
        ms, md, degG, mxw, magg, EM * 32u);
    bn_relu_k<128><<<cdivu(G128, T), T, 0, stream>>>(magg, g2, be2, m2, v2, glob, G128);

    // --- hc = gcn(glob, MW3, mb3) ------------------------------------------
    gemm_wmma_k<128, 16, 10, false, false><<<cdivu(G, 128), T, 0, stream>>>(
        glob, MW3, nullptr, nullptr, hxw, G);
    init_agg_k<10><<<cdivu(G10, T), T, 0, stream>>>(hxw, degG, mb3, hc, G10);
    scatter_edges1_k<10><<<cdivu(EM * 10u, T), T, 0, stream>>>(
        ms, md, degG, hxw, hc, EM * 10u);

    // --- trailing scalar 0.0 ----------------------------------------------
    fill_k<<<1, 64, 0, stream>>>(out + (size_t)276 * G, 0.0f, 1u);
}
